// MultiHeadSelfAttention_45964740002235
// MI455X (gfx1250) — matrix operand order repaired and checked by reading, not compiled, not run on hardware
//
#include <hip/hip_runtime.h>

// ---------------------------------------------------------------------------
// MI455X (gfx1250) multi-head self-attention, f16 WMMA compute / f32 accum.
// ---------------------------------------------------------------------------

using half_t = _Float16;
typedef __attribute__((ext_vector_type(16))) _Float16 v16h;
typedef __attribute__((ext_vector_type(8)))  _Float16 v8h;
typedef __attribute__((ext_vector_type(8)))  float    v8f;

#define DEVINL static __device__ __forceinline__

constexpr int BATCH   = 2;
constexpr int SEQ     = 2048;
constexpr int DMODEL  = 1024;
constexpr int NHEADS  = 16;
constexpr int HDIM    = 64;      // DMODEL / NHEADS
constexpr int THREEC  = 3 * DMODEL;
constexpr int MTOK    = BATCH * SEQ;   // 4096 tokens
constexpr float LOG2E = 1.4426950408889634f;

DEVINL v8f wmma_f16(v16h a, v16h b, v8f c) {
  // D = A(16x32 f16) * B(32x16 f16) + C(16x16 f32)
  return __builtin_amdgcn_wmma_f32_16x16x32_f16(
      /*neg_a=*/false, a, /*neg_b=*/false, b,
      /*c_mod=*/(short)0, c, /*reuse_a=*/false, /*reuse_b=*/false);
}

union U16H { v16h v; v8h p[2]; };

// A-fragment (16x32, M x K), contiguous K order (matches B):
//   lanes 0-15 : row = lane,    halves = K[k0..k0+15]
//   lanes16-31 : row = lane-16, halves = K[k0+16..k0+31]
DEVINL v16h load_a_frag(const half_t* row0, int ldk, int k0, int lane) {
  const int r = lane & 15, g = lane >> 4;
  const half_t* p = row0 + r * ldk + k0 + 16 * g;
  U16H u;
  u.p[0] = *(const v8h*)(p);
  u.p[1] = *(const v8h*)(p + 8);
  return u.v;
}

// B-fragment (32x16, K x N) from a K-contiguous (N-major) matrix:
//   lanes 0-15 : col N = lane,    halves = K[k0..k0+15]
//   lanes16-31 : col N = lane-16, halves = K[k0+16..k0+31]
DEVINL v16h load_b_frag(const half_t* n0row, int ldn, int k0, int lane) {
  const int n = lane & 15, g = lane >> 4;
  return *(const v16h*)(n0row + (size_t)n * ldn + k0 + 16 * g);
}

DEVINL void wave_lds_fence() {
  __builtin_amdgcn_wave_barrier();
  asm volatile("s_wait_dscnt 0x0" ::: "memory");
  __builtin_amdgcn_wave_barrier();
}

// Async global -> LDS copy (gfx1250 ASYNCcnt path). 16 bytes per lane.
// LDS offset = low 32 bits of the flat pointer (aperture maps addr[31:0]).
DEVINL void async_ld_b128(half_t* lds, const half_t* g) {
  const uint32_t l = (uint32_t)(uintptr_t)lds;
  const uint64_t a = (uint64_t)(uintptr_t)g;
  asm volatile("global_load_async_to_lds_b128 %0, %1, off"
               :: "v"(l), "v"(a) : "memory");
}

DEVINL void wait_asynccnt0() {
  asm volatile("s_wait_asynccnt 0x0" ::: "memory");
}

// ---------------------------------------------------------------------------
// Converts
// ---------------------------------------------------------------------------
__global__ void cvt_f32_f16_kernel(const float* __restrict__ in,
                                   half_t* __restrict__ out, int n) {
  for (int i = blockIdx.x * blockDim.x + threadIdx.x; i < n;
       i += gridDim.x * blockDim.x)
    out[i] = (half_t)in[i];
}

// in: [K][N] f32 row-major  ->  out: [N][K] f16 (K contiguous per column)
__global__ void transpose_f32_f16_kernel(const float* __restrict__ in,
                                         half_t* __restrict__ out,
                                         int K, int N) {
  const size_t total = (size_t)K * N;
  for (size_t i = blockIdx.x * (size_t)blockDim.x + threadIdx.x; i < total;
       i += (size_t)gridDim.x * blockDim.x) {
    int k = (int)(i / N), n = (int)(i % N);
    out[(size_t)n * K + k] = (half_t)in[i];
  }
}

// ---------------------------------------------------------------------------
// Tiled WMMA GEMM:  C[M,N] = A[M,K](f16) * Bt[N,K](f16, K-contig) + bias
// Block = 128 threads (4 waves); block tile 128x64; wave tile 32x64
// (2 A-frags reuse the 4 B-frags -> 8 WMMA per K-step per wave).
// ---------------------------------------------------------------------------
template <bool F16OUT>
__global__ __launch_bounds__(128) void gemm_wmma_kernel(
    const half_t* __restrict__ A, const half_t* __restrict__ Bt,
    const float* __restrict__ bias, half_t* __restrict__ outH,
    float* __restrict__ outF, int K, int N, float qscale, int qcols) {
  const int lane = threadIdx.x & 31;
  const int wave = threadIdx.x >> 5;
  const int m0 = blockIdx.y * 128 + wave * 32;
  const int n0 = blockIdx.x * 64;

  v8f acc[2][4];
  const v8f z = {};
#pragma unroll
  for (int i = 0; i < 2; ++i)
#pragma unroll
    for (int j = 0; j < 4; ++j) acc[i][j] = z;

  const half_t* Arow0 = A + (size_t)m0 * K;
  const half_t* Arow1 = A + (size_t)(m0 + 16) * K;
  for (int k0 = 0; k0 < K; k0 += 32) {
    const v16h a0 = load_a_frag(Arow0, K, k0, lane);
    const v16h a1 = load_a_frag(Arow1, K, k0, lane);
#pragma unroll
    for (int j = 0; j < 4; ++j) {
      const v16h b = load_b_frag(Bt + (size_t)(n0 + 16 * j) * K, K, k0, lane);
      acc[0][j] = wmma_f16(a0, b, acc[0][j]);
      acc[1][j] = wmma_f16(a1, b, acc[1][j]);
    }
  }

  const int cg = lane >> 4, cn = lane & 15;
#pragma unroll
  for (int i = 0; i < 2; ++i) {
#pragma unroll
    for (int j = 0; j < 4; ++j) {
      const int col = n0 + 16 * j + cn;
      const float bv = bias[col];
#pragma unroll
      for (int vr = 0; vr < 8; ++vr) {
        const int row = m0 + 16 * i + vr + 8 * cg;
        float val = acc[i][j][vr] + bv;
        if (F16OUT) {
          if (col < qcols) val *= qscale;
          outH[(size_t)row * N + col] = (half_t)val;
        } else {
          outF[(size_t)row * N + col] = val;
        }
      }
    }
  }
}

// ---------------------------------------------------------------------------
// Flash attention. Grid: (SEQ/64, BATCH*NHEADS). Block: 128 threads (4 waves).
// Each wave owns 16 query rows; Q resident in registers; loops over keys in
// chunks of 32.  K chunk staged in LDS via async copy (shared by all waves);
// V chunk staged transposed; P transposed per-wave through LDS.
// ---------------------------------------------------------------------------
constexpr int VT_LD = 48;  // Vt row stride in halves (conflict padding)

__global__ __launch_bounds__(128) void attn_kernel(
    const half_t* __restrict__ qkv, half_t* __restrict__ attn) {
  __shared__ alignas(64) half_t Klds[32 * HDIM];       // K chunk, row-major
  __shared__ alignas(64) half_t Vt[HDIM * VT_LD];      // V chunk, transposed
  __shared__ alignas(64) half_t Pbuf[4][16 * 32];      // per-wave P transpose

  const int lane = threadIdx.x & 31;
  const int wave = threadIdx.x >> 5;
  const int bh = blockIdx.y;
  const int b = bh / NHEADS, h = bh % NHEADS;
  const int bt0 = b * SEQ;
  const int m0 = blockIdx.x * 64 + wave * 16;  // query row (within batch)
  const int g = lane >> 4, cn = lane & 15;

  // Q fragments: dims 0..31 and 32..63 (Q already scaled by 1/sqrt(Dh))
  const half_t* qbase = qkv + (size_t)(bt0 + m0) * THREEC + h * HDIM;
  const v16h aq0 = load_a_frag(qbase, THREEC, 0, lane);
  const v16h aq1 = load_a_frag(qbase, THREEC, 32, lane);

  float m_run[8], l_run[8];
  v8f o[4];
  const v8f z = {};
#pragma unroll
  for (int vr = 0; vr < 8; ++vr) { m_run[vr] = -1.0e30f; l_run[vr] = 0.0f; }
#pragma unroll
  for (int j = 0; j < 4; ++j) o[j] = z;

  half_t* P = &Pbuf[wave][0];
  const int skey = threadIdx.x >> 3;        // 0..15 (staging key id)
  const int sd8 = (threadIdx.x & 7) * 8;    // 0..56 (staging dim offset)

  for (int c0 = 0; c0 < SEQ; c0 += 32) {
    __syncthreads();  // protect previous iteration's Klds/Vt reads

    // ---- async-stage K chunk (row-major [key][d]) into LDS ----
    const half_t* kglob =
        qkv + (size_t)(bt0 + c0) * THREEC + DMODEL + h * HDIM;
    async_ld_b128(&Klds[skey * HDIM + sd8],
                  kglob + (size_t)skey * THREEC + sd8);
    async_ld_b128(&Klds[(skey + 16) * HDIM + sd8],
                  kglob + (size_t)(skey + 16) * THREEC + sd8);

    // ---- stage V chunk transposed into LDS: Vt[d][key] ----
#pragma unroll
    for (int i = 0; i < 16; ++i) {
      const int idx = threadIdx.x + 128 * i;  // 32 keys x 64 dims
      const int kk = idx >> 6, d = idx & 63;
      Vt[d * VT_LD + kk] =
          qkv[(size_t)(bt0 + c0 + kk) * THREEC + 2 * DMODEL + h * HDIM + d];
    }
    wait_asynccnt0();
    __syncthreads();

    // ---- S = Q * K^T for 32 keys (two 16-col tiles), K from LDS ----
    v8f s0 = z, s1 = z;
    s0 = wmma_f16(aq0, load_b_frag(&Klds[0], HDIM, 0, lane), s0);
    s0 = wmma_f16(aq1, load_b_frag(&Klds[0], HDIM, 32, lane), s0);
    s1 = wmma_f16(aq0, load_b_frag(&Klds[16 * HDIM], HDIM, 0, lane), s1);
    s1 = wmma_f16(aq1, load_b_frag(&Klds[16 * HDIM], HDIM, 32, lane), s1);

    // ---- online softmax (row = C-layout row, spread over 16 lanes) ----
    float alpha[8];
#pragma unroll
    for (int vr = 0; vr < 8; ++vr) {
      float mx = fmaxf(s0[vr], s1[vr]);
      mx = fmaxf(mx, __shfl_xor(mx, 1, 32));
      mx = fmaxf(mx, __shfl_xor(mx, 2, 32));
      mx = fmaxf(mx, __shfl_xor(mx, 4, 32));
      mx = fmaxf(mx, __shfl_xor(mx, 8, 32));
      const float mn = fmaxf(m_run[vr], mx);
      alpha[vr] = exp2f((m_run[vr] - mn) * LOG2E);
      const float p0 = exp2f((s0[vr] - mn) * LOG2E);
      const float p1 = exp2f((s1[vr] - mn) * LOG2E);
      float rs = p0 + p1;
      rs += __shfl_xor(rs, 1, 32);
      rs += __shfl_xor(rs, 2, 32);
      rs += __shfl_xor(rs, 4, 32);
      rs += __shfl_xor(rs, 8, 32);
      l_run[vr] = l_run[vr] * alpha[vr] + rs;
      m_run[vr] = mn;
      s0[vr] = p0;
      s1[vr] = p1;
    }
#pragma unroll
    for (int j = 0; j < 4; ++j)
#pragma unroll
      for (int vr = 0; vr < 8; ++vr) o[j][vr] *= alpha[vr];

    // ---- P: C-layout f32 -> A-layout f16 via per-wave LDS transpose ----
#pragma unroll
    for (int vr = 0; vr < 8; ++vr) {
      const int row = vr + 8 * g;
      P[row * 32 + cn] = (half_t)s0[vr];
      P[row * 32 + 16 + cn] = (half_t)s1[vr];
    }
    wave_lds_fence();
    const v16h ap = load_a_frag(P, 32, 0, lane);

    // ---- O += P(16x32) * V(32x64) ----
#pragma unroll
    for (int j = 0; j < 4; ++j) {
      const v16h bv = *(const v16h*)(&Vt[(16 * j + cn) * VT_LD + 16 * g]);
      o[j] = wmma_f16(ap, bv, o[j]);
    }
  }

  // ---- finalize: divide by l, store f16 in [token][h*64+d] layout ----
#pragma unroll
  for (int vr = 0; vr < 8; ++vr) {
    const float inv_l = 1.0f / l_run[vr];
    const int tok = m0 + vr + 8 * g;
#pragma unroll
    for (int j = 0; j < 4; ++j) {
      const int d = 16 * j + cn;
      attn[(size_t)(bt0 + tok) * DMODEL + h * HDIM + d] =
          (half_t)(o[j][vr] * inv_l);
    }
  }
}

// ---------------------------------------------------------------------------
// Launch
// ---------------------------------------------------------------------------
extern "C" void kernel_launch(void* const* d_in, const int* in_sizes, int n_in,
                              void* d_out, int out_size, void* d_ws,
                              size_t ws_size, hipStream_t stream) {
  const float* x     = (const float*)d_in[0];  // [B,T,C]
  const float* w_qkv = (const float*)d_in[1];  // [C, 3C]
  const float* b_qkv = (const float*)d_in[2];  // [3C]
  const float* w_out = (const float*)d_in[3];  // [C, C]
  const float* b_out = (const float*)d_in[4];  // [C]
  float* out = (float*)d_out;                  // [B,T,C] f32

  unsigned char* ws = (unsigned char*)d_ws;
  size_t off = 0;
  half_t* xh = (half_t*)(ws + off);      off += (size_t)MTOK * DMODEL * 2;
  half_t* wqkvT = (half_t*)(ws + off);   off += (size_t)THREEC * DMODEL * 2;
  half_t* woutT = (half_t*)(ws + off);   off += (size_t)DMODEL * DMODEL * 2;
  half_t* qkvh = (half_t*)(ws + off);    off += (size_t)MTOK * THREEC * 2;
  half_t* attnh = (half_t*)(ws + off);   off += (size_t)MTOK * DMODEL * 2;
  (void)ws_size; (void)in_sizes; (void)n_in; (void)out_size;

  // 1) convert inputs to f16 (weights transposed to K-contiguous layout)
  cvt_f32_f16_kernel<<<dim3(1024), dim3(256), 0, stream>>>(
      x, xh, MTOK * DMODEL);
  transpose_f32_f16_kernel<<<dim3(2048), dim3(256), 0, stream>>>(
      w_qkv, wqkvT, DMODEL, THREEC);
  transpose_f32_f16_kernel<<<dim3(1024), dim3(256), 0, stream>>>(
      w_out, woutT, DMODEL, DMODEL);

  // 2) qkv = x @ w_qkv + b_qkv  (Q cols pre-scaled by 1/sqrt(Dh) = 1/8)
  gemm_wmma_kernel<true>
      <<<dim3(THREEC / 64, MTOK / 128), dim3(128), 0, stream>>>(
          xh, wqkvT, b_qkv, qkvh, nullptr, DMODEL, THREEC, 0.125f, DMODEL);

  // 3) flash attention -> attnh [token][C]
  attn_kernel<<<dim3(SEQ / 64, BATCH * NHEADS), dim3(128), 0, stream>>>(
      qkvh, attnh);

  // 4) out = attn @ w_out + b_out (f32 store)
  gemm_wmma_kernel<false>
      <<<dim3(DMODEL / 64, MTOK / 128), dim3(128), 0, stream>>>(
          attnh, woutT, b_out, nullptr, out, DMODEL, DMODEL, 1.0f, 0);
}
